// Block_74766790689102
// MI455X (gfx1250) — compile-verified, matrix operand order checked by reading, not run on hardware
//
#include <hip/hip_runtime.h>
#include <math.h>

typedef __attribute__((ext_vector_type(16))) __bf16 v16bf;
typedef __attribute__((ext_vector_type(8)))  float  v8f;
typedef __attribute__((ext_vector_type(4)))  unsigned int u32x4;
typedef __attribute__((ext_vector_type(8)))  unsigned int u32x8;

#define DIM      1024
#define HEADS    16
#define HEAD_DIM 64
#define HIDDEN   4096
#define SEQ      2048
#define BATCH    2
#define ROWS     (BATCH * SEQ)   // 4096 tokens
#define QK_SCALE 0.125f          // 64^-0.5

// ---------------------------------------------------------------------------
// WMMA helpers
// ---------------------------------------------------------------------------
__device__ __forceinline__ v8f wmma_bf16(v16bf a, v16bf b, v8f c) {
  // D(16x16,f32) = A(16x32,bf16) * B(32x16,bf16) + C
  return __builtin_amdgcn_wmma_f32_16x16x32_bf16(false, a, false, b, (short)0, c,
                                                 false, false);
}

// 16x32 bf16 A-fragment (== 32x16 B-fragment layout) where each of the 16
// rows/cols is K-contiguous with `stride` elements. CDNA5 ISA layout:
// lane r<16 holds K=0..7 (v0..3) and 16..23 (v4..7); lane r+16 holds K=8..15
// and 24..31 -> two 16B chunks per lane.
__device__ __forceinline__ v16bf load_frag(const __bf16* base, int stride, int lane) {
  const int r  = lane & 15;
  const int hi = lane >> 4;
  const __bf16* p = base + r * stride + hi * 8;
  union { uint4 u[2]; v16bf v; } u;
  u.u[0] = *(const uint4*)(p);
  u.u[1] = *(const uint4*)(p + 16);
  return u.v;
}

// Swap data between lane halves (lane <-> lane^16) in one DS op:
// ds_swizzle group-of-32 mode, xor_mask=0x10, and_mask=0x1f -> imm 0x401F.
__device__ __forceinline__ float swap16(float x) {
  int v = __builtin_amdgcn_ds_swizzle(__builtin_bit_cast(int, x), 0x401F);
  return __builtin_bit_cast(float, v);
}

// ---------------------------------------------------------------------------
// Tensor Data Mover: DMA a 2-D tile (rows x cols of bf16, row pitch stride0)
// into LDS with per-row padding. Descriptor per CDNA5 ISA
// (group0 + group1, 2-D tensor => VADDR2/VADDR3 omitted = NULL).
// pad codes: interval 2^(c+1) DWORDs of data, amount (c+1) DWORDs inserted.
// Tracked on TENSORcnt; caller waits with s_wait_tensorcnt.
// ---------------------------------------------------------------------------
__device__ __forceinline__ void tdm_load_2d(unsigned lds_off, const __bf16* gsrc,
                                            unsigned tensor_d0, unsigned tensor_d1,
                                            unsigned stride0_elems,
                                            unsigned tile_d0, unsigned tile_d1,
                                            unsigned pad_interval_code,
                                            unsigned pad_amount_code) {
  unsigned long long ga = (unsigned long long)(uintptr_t)gsrc;
  u32x4 g0;
  g0[0] = 1u;                                   // count=1 valid descriptor
  g0[1] = lds_off;                              // LDS byte address
  g0[2] = (unsigned)ga;                         // global_addr[31:0]
  g0[3] = (unsigned)((ga >> 32) & 0x1FFFFFFu)   // global_addr[56:32]
          | 0x80000000u;                        // type=2 ("image")
  u32x8 g1;
  g1[0] = (1u << 16)                            // data_size = 2 bytes
          | (1u << 20)                          // pad_enable
          | (pad_interval_code << 22)
          | (pad_amount_code << 25);
  g1[1] = (tensor_d0 & 0xFFFFu) << 16;                           // dim0[15:0]
  g1[2] = ((tensor_d0 >> 16) & 0xFFFFu) | ((tensor_d1 & 0xFFFFu) << 16);
  g1[3] = ((tensor_d1 >> 16) & 0xFFFFu) | ((tile_d0 & 0xFFFFu) << 16);
  g1[4] = tile_d1 & 0xFFFFu;                    // tile_dim1 (tile_dim2 = 0)
  g1[5] = stride0_elems;                        // tensor_dim0_stride[31:0]
  g1[6] = 0u;                                   // stride0[47:32], stride1[15:0]
  g1[7] = 0u;
  asm volatile("tensor_load_to_lds %0, %1" :: "s"(g0), "s"(g1) : "memory");
}

// ---------------------------------------------------------------------------
// fp32 -> bf16 weight cast (weights reused by all GEMMs)
// ---------------------------------------------------------------------------
__global__ __launch_bounds__(256) void cast_bf16_kernel(const float* __restrict__ src,
                                                        __bf16* __restrict__ dst, int n) {
  int i = (blockIdx.x * 256 + threadIdx.x) * 4;
  if (i < n) {
    float4 v = *(const float4*)(src + i);
    dst[i + 0] = (__bf16)v.x;
    dst[i + 1] = (__bf16)v.y;
    dst[i + 2] = (__bf16)v.z;
    dst[i + 3] = (__bf16)v.w;
  }
}

// ---------------------------------------------------------------------------
// LayerNorm over C=1024, one token per 256-thread block, bf16 output
// ---------------------------------------------------------------------------
__global__ __launch_bounds__(256) void ln_bf16_kernel(const float* __restrict__ x,
                                                      const float* __restrict__ g,
                                                      const float* __restrict__ b,
                                                      __bf16* __restrict__ out) {
  const int row = blockIdx.x;
  const int t   = threadIdx.x;
  float4 v = *(const float4*)(x + (size_t)row * DIM + 4 * t);
  float s = v.x + v.y + v.z + v.w;
  float q = v.x * v.x + v.y * v.y + v.z * v.z + v.w * v.w;
#pragma unroll
  for (int m = 1; m < 32; m <<= 1) {
    s += __shfl_xor(s, m, 32);
    q += __shfl_xor(q, m, 32);
  }
  __shared__ float ss[8], qq[8];
  if ((t & 31) == 0) { ss[t >> 5] = s; qq[t >> 5] = q; }
  __syncthreads();
  s = 0.f; q = 0.f;
#pragma unroll
  for (int i = 0; i < 8; i++) { s += ss[i]; q += qq[i]; }
  const float mu   = s * (1.f / DIM);
  const float var  = q * (1.f / DIM) - mu * mu;
  const float rinv = rsqrtf(var + 1e-5f);
  float4 gv = *(const float4*)(g + 4 * t);
  float4 bv = *(const float4*)(b + 4 * t);
  __bf16* o = out + (size_t)row * DIM + 4 * t;
  o[0] = (__bf16)((v.x - mu) * rinv * gv.x + bv.x);
  o[1] = (__bf16)((v.y - mu) * rinv * gv.y + bv.y);
  o[2] = (__bf16)((v.z - mu) * rinv * gv.z + bv.z);
  o[3] = (__bf16)((v.w - mu) * rinv * gv.w + bv.w);
}

// ---------------------------------------------------------------------------
// GEMM: out[M,N] = A[M,K](bf16,row-major) * W[N,K](bf16,row-major)^T
// 128x128 block tile, 8 waves each 32x64. K staged 64-deep via TDM into
// double-buffered LDS (padded row stride 72 bf16 => conflict-free
// ds_read_b128 fragment loads). Only wave 0 drives the DMA; s_wait_tensorcnt
// overlaps next slab's DMA with current slab's 16 WMMAs/wave.
//   MODE 0: QKV scatter (q scaled, k, v transposed to [b,h,d,n]) bf16
//   MODE 1: + bias + residual -> fp32
//   MODE 2: + bias + exact GELU -> bf16
// ---------------------------------------------------------------------------
#define KSLAB   64
#define LDSTR   72   // 64 data + 8 pad elements per row

template <int MODE>
__global__ __launch_bounds__(256) void gemm_bf16_kernel(
    const __bf16* __restrict__ A, const __bf16* __restrict__ W,
    int M, int N, int K,
    const float* __restrict__ bias, const float* __restrict__ resid,
    float* __restrict__ outf, __bf16* __restrict__ outbf,
    __bf16* __restrict__ outq, __bf16* __restrict__ outk, __bf16* __restrict__ outvT) {
  __shared__ __align__(16) __bf16 As[2][128 * LDSTR];
  __shared__ __align__(16) __bf16 Ws[2][128 * LDSTR];

  const int t = threadIdx.x;
  const int lane = t & 31, wave = t >> 5;
  const int wm = wave >> 1, wn = wave & 1;        // 4x2 wave grid
  const int m0 = blockIdx.y * 128, n0 = blockIdx.x * 128;

  v8f acc[2][4];
#pragma unroll
  for (int mi = 0; mi < 2; mi++)
#pragma unroll
    for (int ni = 0; ni < 4; ni++) acc[mi][ni] = {};

  if (wave == 0) {  // prologue DMA: slab 0 -> buffer 0
    tdm_load_2d((unsigned)(uintptr_t)&As[0][0], A + (size_t)m0 * K, K, M, K,
                KSLAB, 128, 4, 3);
    tdm_load_2d((unsigned)(uintptr_t)&Ws[0][0], W + (size_t)n0 * K, K, N, K,
                KSLAB, 128, 4, 3);
  }

  for (int k0 = 0; k0 < K; k0 += KSLAB) {
    const int cur = (k0 / KSLAB) & 1;
    if (wave == 0) {
      if (k0 + KSLAB < K) {
        tdm_load_2d((unsigned)(uintptr_t)&As[cur ^ 1][0],
                    A + (size_t)m0 * K + (k0 + KSLAB), K, M, K, KSLAB, 128, 4, 3);
        tdm_load_2d((unsigned)(uintptr_t)&Ws[cur ^ 1][0],
                    W + (size_t)n0 * K + (k0 + KSLAB), K, N, K, KSLAB, 128, 4, 3);
        __builtin_amdgcn_s_wait_tensorcnt(2);  // current slab landed
      } else {
        __builtin_amdgcn_s_wait_tensorcnt(0);
      }
    }
    __syncthreads();

    const __bf16* Ab = &As[cur][0];
    const __bf16* Wb = &Ws[cur][0];
#pragma unroll
    for (int kk = 0; kk < KSLAB; kk += 32) {
      v16bf af[2], wf[4];
#pragma unroll
      for (int mi = 0; mi < 2; mi++)
        af[mi] = load_frag(Ab + (wm * 32 + mi * 16) * LDSTR + kk, LDSTR, lane);
#pragma unroll
      for (int ni = 0; ni < 4; ni++)
        wf[ni] = load_frag(Wb + (wn * 64 + ni * 16) * LDSTR + kk, LDSTR, lane);
#pragma unroll
      for (int mi = 0; mi < 2; mi++)
#pragma unroll
        for (int ni = 0; ni < 4; ni++)
          acc[mi][ni] = wmma_bf16(af[mi], wf[ni], acc[mi][ni]);
    }
    __syncthreads();  // tile consumed; safe for DMA to overwrite next round
  }

  // Epilogue. D layout: lane holds col = lane&15; VGPR i holds row i + 8*(lane>>4).
  const int hi = lane >> 4, lc = lane & 15;
#pragma unroll
  for (int mi = 0; mi < 2; mi++) {
#pragma unroll
    for (int ni = 0; ni < 4; ni++) {
      const int gr0 = m0 + wm * 32 + mi * 16 + 8 * hi;
      const int gc  = n0 + wn * 64 + ni * 16 + lc;
#pragma unroll
      for (int i = 0; i < 8; i++) {
        const int gr = gr0 + i;
        float val = acc[mi][ni][i];
        if constexpr (MODE == 0) {
          const int which = gc >> 10;
          const int hh    = (gc >> 6) & (HEADS - 1);
          const int d     = gc & (HEAD_DIM - 1);
          const int bb    = gr >> 11;
          const int n     = gr & (SEQ - 1);
          const size_t bh = (size_t)(bb * HEADS + hh);
          if (which == 0)
            outq[(bh * SEQ + n) * HEAD_DIM + d] = (__bf16)(val * QK_SCALE);
          else if (which == 1)
            outk[(bh * SEQ + n) * HEAD_DIM + d] = (__bf16)val;
          else
            outvT[(bh * HEAD_DIM + d) * SEQ + n] = (__bf16)val;  // V transposed
        } else if constexpr (MODE == 1) {
          outf[(size_t)gr * N + gc] = val + bias[gc] + resid[(size_t)gr * N + gc];
        } else {
          float xg = val + bias[gc];
          outbf[(size_t)gr * N + gc] =
              (__bf16)(0.5f * xg * (1.f + erff(xg * 0.70710678118654752f)));
        }
      }
    }
  }
}

// ---------------------------------------------------------------------------
// Flash attention: 128 queries per block (16 per wave). K/V^T tiles for each
// 32-key step are TDM-DMA'd into double-buffered LDS and shared by all 8
// waves (8x cut in global traffic vs per-wave loads). Computes S^T = K*Q^T so
// the softmax'd P tile is already in B-fragment layout for O^T = V^T * P^T.
// All fragments for a step are loaded into independent registers up front so
// the ds_load_b128s clause together and overlap with the WMMAs.
// ---------------------------------------------------------------------------
__global__ __launch_bounds__(256) void flash_kernel(const __bf16* __restrict__ Q,
                                                    const __bf16* __restrict__ Km,
                                                    const __bf16* __restrict__ VT,
                                                    __bf16* __restrict__ O) {
  __shared__ __align__(16) __bf16 Ks[2][32 * 72];  // 32 keys x 64 d, stride 72
  __shared__ __align__(16) __bf16 Vs[2][64 * 40];  // 64 d x 32 keys, stride 40

  const int t = threadIdx.x, lane = t & 31, wave = t >> 5;
  const int bh = blockIdx.y;                      // b*16 + h
  const int b  = bh >> 4, h = bh & 15;
  const int n0 = blockIdx.x * 128 + wave * 16;    // query tile base
  const __bf16* Qb = Q  + (size_t)bh * SEQ * HEAD_DIM;
  const __bf16* Kb = Km + (size_t)bh * SEQ * HEAD_DIM;
  const __bf16* Vb = VT + (size_t)bh * HEAD_DIM * SEQ;

  // Q^T B-fragments (lane = query, chunks over d); Q pre-scaled by 1/sqrt(d).
  v16bf qf0 = load_frag(Qb + (size_t)n0 * HEAD_DIM,      HEAD_DIM, lane);
  v16bf qf1 = load_frag(Qb + (size_t)n0 * HEAD_DIM + 32, HEAD_DIM, lane);

  v8f oacc[4];
#pragma unroll
  for (int dt = 0; dt < 4; dt++) oacc[dt] = {};
  float m_run = -3.0e38f, l_run = 0.f;

  if (wave == 0) {  // prologue DMA: keys 0..31 -> buffer 0
    tdm_load_2d((unsigned)(uintptr_t)&Ks[0][0], Kb, HEAD_DIM, SEQ, HEAD_DIM,
                HEAD_DIM, 32, 4, 3);
    tdm_load_2d((unsigned)(uintptr_t)&Vs[0][0], Vb, SEQ, HEAD_DIM, SEQ,
                32, HEAD_DIM, 3, 3);
  }

  for (int j = 0; j < SEQ; j += 32) {
    const int cur = (j >> 5) & 1;
    if (wave == 0) {
      if (j + 32 < SEQ) {
        tdm_load_2d((unsigned)(uintptr_t)&Ks[cur ^ 1][0],
                    Kb + (size_t)(j + 32) * HEAD_DIM, HEAD_DIM, SEQ, HEAD_DIM,
                    HEAD_DIM, 32, 4, 3);
        tdm_load_2d((unsigned)(uintptr_t)&Vs[cur ^ 1][0],
                    Vb + (j + 32), SEQ, HEAD_DIM, SEQ, 32, HEAD_DIM, 3, 3);
        __builtin_amdgcn_s_wait_tensorcnt(2);
      } else {
        __builtin_amdgcn_s_wait_tensorcnt(0);
      }
    }
    __syncthreads();
    const __bf16* Kt = &Ks[cur][0];
    const __bf16* Vt = &Vs[cur][0];

    // Load all K fragments and V fragments into independent registers first.
    v16bf kf0 = load_frag(Kt,                72, lane);
    v16bf kf1 = load_frag(Kt + 32,           72, lane);
    v16bf kf2 = load_frag(Kt + 16 * 72,      72, lane);
    v16bf kf3 = load_frag(Kt + 16 * 72 + 32, 72, lane);
    v16bf vf0 = load_frag(Vt,                40, lane);
    v16bf vf1 = load_frag(Vt + 16 * 40,      40, lane);
    v16bf vf2 = load_frag(Vt + 32 * 40,      40, lane);
    v16bf vf3 = load_frag(Vt + 48 * 40,      40, lane);

    // S^T tiles: rows = keys (tile0: j..j+15, tile1: j+16..j+31), cols = queries
    v8f s0 = {}, s1 = {};
    s0 = wmma_bf16(kf0, qf0, s0);
    s1 = wmma_bf16(kf2, qf0, s1);
    s0 = wmma_bf16(kf1, qf1, s0);
    s1 = wmma_bf16(kf3, qf1, s1);

    // Online softmax per query (lane pair {q, q+16} covers all 32 keys).
    float sm = -3.0e38f;
#pragma unroll
    for (int i = 0; i < 8; i++) sm = fmaxf(sm, fmaxf(s0[i], s1[i]));
    sm = fmaxf(sm, swap16(sm));
    const float m_new = fmaxf(m_run, sm);
    const float corr  = __expf(m_run - m_new);
    float rs = 0.f;
    v16bf pf;  // P^T in B-fragment layout: elems 0..7 = s0 keys, 8..15 = s1 keys
#pragma unroll
    for (int i = 0; i < 8; i++) {
      float p0 = __expf(s0[i] - m_new);
      float p1 = __expf(s1[i] - m_new);
      rs += p0 + p1;
      pf[i]     = (__bf16)p0;
      pf[8 + i] = (__bf16)p1;
    }
    rs += swap16(rs);
    l_run = l_run * corr + rs;
    m_run = m_new;
#pragma unroll
    for (int dt = 0; dt < 4; dt++)
#pragma unroll
      for (int i = 0; i < 8; i++) oacc[dt][i] *= corr;
    // O^T += V^T * P^T  (A-frag rows = d, K-contiguous over keys, from LDS)
    oacc[0] = wmma_bf16(vf0, pf, oacc[0]);
    oacc[1] = wmma_bf16(vf1, pf, oacc[1]);
    oacc[2] = wmma_bf16(vf2, pf, oacc[2]);
    oacc[3] = wmma_bf16(vf3, pf, oacc[3]);
    __syncthreads();  // tile consumed; DMA may overwrite next round
  }

  const float inv = 1.f / l_run;
  const int hi = lane >> 4, qc = lane & 15;
  __bf16* orow = O + (size_t)(b * SEQ + n0 + qc) * DIM + h * HEAD_DIM;
#pragma unroll
  for (int dt = 0; dt < 4; dt++)
#pragma unroll
    for (int i = 0; i < 8; i++)
      orow[dt * 16 + i + 8 * hi] = (__bf16)(oacc[dt][i] * inv);
}

// ---------------------------------------------------------------------------
// Host launcher
// ---------------------------------------------------------------------------
extern "C" void kernel_launch(void* const* d_in, const int* in_sizes, int n_in,
                              void* d_out, int out_size, void* d_ws, size_t ws_size,
                              hipStream_t stream) {
  const float* x      = (const float*)d_in[0];
  const float* ln1_g  = (const float*)d_in[1];
  const float* ln1_b  = (const float*)d_in[2];
  const float* w_qkv  = (const float*)d_in[3];
  const float* w_proj = (const float*)d_in[4];
  const float* b_proj = (const float*)d_in[5];
  const float* ln2_g  = (const float*)d_in[6];
  const float* ln2_b  = (const float*)d_in[7];
  const float* w_fc1  = (const float*)d_in[8];
  const float* b_fc1  = (const float*)d_in[9];
  const float* w_fc2  = (const float*)d_in[10];
  const float* b_fc2  = (const float*)d_in[11];

  char* ws = (char*)d_ws;
  // Workspace layout (bytes). act_bf aliases q/k/vT/o which are dead by then.
  __bf16* wqkv_bf  = (__bf16*)(ws + 0);          //  6 MB
  __bf16* wproj_bf = (__bf16*)(ws + 6291456);    //  2 MB
  __bf16* wfc1_bf  = (__bf16*)(ws + 8388608);    //  8 MB
  __bf16* wfc2_bf  = (__bf16*)(ws + 16777216);   //  8 MB
  __bf16* h_bf     = (__bf16*)(ws + 25165824);   //  8 MB (LN1 out, reused for LN2 out)
  __bf16* q_bf     = (__bf16*)(ws + 33554432);   //  8 MB
  __bf16* k_bf     = (__bf16*)(ws + 41943040);   //  8 MB
  __bf16* vT_bf    = (__bf16*)(ws + 50331648);   //  8 MB
  __bf16* o_bf     = (__bf16*)(ws + 58720256);   //  8 MB
  float*  x1       = (float*)(ws + 67108864);    // 16 MB (attn residual sum)
  __bf16* act_bf   = (__bf16*)(ws + 33554432);   // 32 MB alias (GELU activations)

  // 1. Weights -> bf16
  cast_bf16_kernel<<<(3 * DIM * DIM) / 1024, 256, 0, stream>>>(w_qkv, wqkv_bf, 3 * DIM * DIM);
  cast_bf16_kernel<<<(DIM * DIM) / 1024, 256, 0, stream>>>(w_proj, wproj_bf, DIM * DIM);
  cast_bf16_kernel<<<(HIDDEN * DIM) / 1024, 256, 0, stream>>>(w_fc1, wfc1_bf, HIDDEN * DIM);
  cast_bf16_kernel<<<(DIM * HIDDEN) / 1024, 256, 0, stream>>>(w_fc2, wfc2_bf, DIM * HIDDEN);

  // 2. LN1
  ln_bf16_kernel<<<ROWS, 256, 0, stream>>>(x, ln1_g, ln1_b, h_bf);

  // 3. QKV GEMM, scatter q (scaled) / k / v^T
  gemm_bf16_kernel<0><<<dim3((3 * DIM) / 128, ROWS / 128), 256, 0, stream>>>(
      h_bf, wqkv_bf, ROWS, 3 * DIM, DIM, nullptr, nullptr, nullptr, nullptr,
      q_bf, k_bf, vT_bf);

  // 4. Flash attention
  flash_kernel<<<dim3(SEQ / 128, BATCH * HEADS), 256, 0, stream>>>(q_bf, k_bf, vT_bf, o_bf);

  // 5. Proj GEMM + bias + residual -> x1 (fp32)
  gemm_bf16_kernel<1><<<dim3(DIM / 128, ROWS / 128), 256, 0, stream>>>(
      o_bf, wproj_bf, ROWS, DIM, DIM, b_proj, x, x1, nullptr, nullptr, nullptr, nullptr);

  // 6. LN2
  ln_bf16_kernel<<<ROWS, 256, 0, stream>>>(x1, ln2_g, ln2_b, h_bf);

  // 7. FC1 GEMM + bias + exact GELU -> bf16 activations
  gemm_bf16_kernel<2><<<dim3(HIDDEN / 128, ROWS / 128), 256, 0, stream>>>(
      h_bf, wfc1_bf, ROWS, HIDDEN, DIM, b_fc1, nullptr, nullptr, act_bf,
      nullptr, nullptr, nullptr);

  // 8. FC2 GEMM + bias + residual -> d_out (fp32)
  gemm_bf16_kernel<1><<<dim3(DIM / 128, ROWS / 128), 256, 0, stream>>>(
      act_bf, wfc2_bf, ROWS, DIM, HIDDEN, b_fc2, x1, (float*)d_out,
      nullptr, nullptr, nullptr, nullptr);
}
